// MoELayer_79731772883260
// MI455X (gfx1250) — matrix operand order repaired and checked by reading, not compiled, not run on hardware
//
#include <hip/hip_runtime.h>
#include <math.h>

// ---- problem constants (fixed by reference) ----
#define DMODEL   1024
#define HID      2730
#define NH2      5460
#define NEXP     8
#define NTOK     8192
// ---- padded layouts for aligned bf16 WMMA fragment loads ----
#define H2OFF    2736               // h2 columns re-based to 16-aligned offset
#define W12STRIDE 5472              // [0,2730)=h1, [2736,5466)=h2, rest zero
#define HPAD     2752               // K of GEMM2 padded to multiple of 32
// ---- tiling ----
#define BM       32                 // tokens per block (2 x 16-row WMMA tiles)
#define TPB      256                // 8 waves of 32
#define NWAVE    8
#define TILES    (NTOK / BM)        // 256
#define NT1      171                // ceil(2730/16) column-pair tiles
#define HCOVER   (NT1 * 16)         // 2736 columns written by GEMM1

typedef __attribute__((ext_vector_type(16))) __bf16   v16bf;
typedef __attribute__((ext_vector_type(8)))  float    v8f;
typedef __attribute__((ext_vector_type(4)))  unsigned u32x4;

union Frag { u32x4 q[2]; v16bf v; };

__device__ __forceinline__ unsigned short f2bf(float f) {
  unsigned u = __float_as_uint(f);
  unsigned r = 0x7FFFu + ((u >> 16) & 1u);          // round-to-nearest-even
  return (unsigned short)((u + r) >> 16);
}

__device__ __forceinline__ v8f vzero8() {
  v8f z = {0.f, 0.f, 0.f, 0.f, 0.f, 0.f, 0.f, 0.f};
  return z;
}

// A-fragment: lane holds row M=(lane&15); lanes<16 take K {0..7,16..23},
// lanes>=16 take K {8..15,24..31}  (ISA 7.12.2, 16-bit A 16x32)
__device__ __forceinline__ v16bf ld_frag_row(const unsigned short* rowbase, int k0, int hi) {
  Frag f;
  f.q[0] = *(const u32x4*)(rowbase + k0 + hi);
  f.q[1] = *(const u32x4*)(rowbase + k0 + hi + 16);
  return f.v;
}

// B-fragment: lane holds column N=(lane&15); K order matches A-frag (hi+{0..7}, hi+16+{0..7})
__device__ __forceinline__ v16bf ld_frag_b(const unsigned short* p, int ld, int hi) {
  Frag f;
  unsigned short t[16];
  #pragma unroll
  for (int s = 0; s < 16; ++s) t[s] = p[(long long)(hi + (s < 8 ? s : s + 8)) * ld];
  __builtin_memcpy(&f, t, 32);
  return f.v;
}

__device__ __forceinline__ v8f wmma_bf16(v16bf a, v16bf b, v8f c) {
  return __builtin_amdgcn_wmma_f32_16x16x32_bf16(false, a, false, b, (short)0, c, false, false);
}

// ------------------------------------------------------------------
__global__ void zero_u32(unsigned* __restrict__ p, long long n) {
  long long stride = (long long)gridDim.x * blockDim.x;
  for (long long i = (long long)blockIdx.x * blockDim.x + threadIdx.x; i < n; i += stride)
    p[i] = 0u;
}

__global__ void conv_x(const float* __restrict__ src, unsigned short* __restrict__ dst, long long n) {
  long long stride = (long long)gridDim.x * blockDim.x;
  for (long long i = (long long)blockIdx.x * blockDim.x + threadIdx.x; i < n; i += stride)
    dst[i] = f2bf(src[i]);
}

// W12 [E][D][5460] f32 -> [E][D][5472] bf16; h2 re-based to col 2736; pads zero
__global__ void conv_w12(const float* __restrict__ src, unsigned short* __restrict__ dst) {
  const long long total = (long long)NEXP * DMODEL * W12STRIDE;
  long long stride = (long long)gridDim.x * blockDim.x;
  for (long long i = (long long)blockIdx.x * blockDim.x + threadIdx.x; i < total; i += stride) {
    int c = (int)(i % W12STRIDE);
    long long ek = i / W12STRIDE;          // e*DMODEL + k
    float v = 0.f;
    if (c < HID)                       v = src[ek * NH2 + c];
    else if (c >= H2OFF && (c - H2OFF) < HID) v = src[ek * NH2 + HID + (c - H2OFF)];
    dst[i] = f2bf(v);
  }
}

// W3 [E][2730][1024] f32 -> [E][2752][1024] bf16, K zero-padded
__global__ void conv_w3(const float* __restrict__ src, unsigned short* __restrict__ dst) {
  const long long total = (long long)NEXP * HPAD * DMODEL;
  long long stride = (long long)gridDim.x * blockDim.x;
  for (long long i = (long long)blockIdx.x * blockDim.x + threadIdx.x; i < total; i += stride) {
    int n = (int)(i % DMODEL);
    long long ek = i / DMODEL;             // e*HPAD + k
    int k = (int)(ek % HPAD);
    int e = (int)(ek / HPAD);
    float v = (k < HID) ? src[((long long)e * HID + k) * DMODEL + n] : 0.f;
    dst[i] = f2bf(v);
  }
}

// routing: one thread per token; top-2 of 8 logits, softmax, append to lists
__global__ void route(const float* __restrict__ x, const float* __restrict__ Wr,
                      int* __restrict__ counts, int* __restrict__ tokl,
                      float* __restrict__ gatl, int T) {
  int t = blockIdx.x * blockDim.x + threadIdx.x;
  if (t >= T) return;
  const float* xr = x + (long long)t * DMODEL;
  float l[NEXP];
  #pragma unroll
  for (int e = 0; e < NEXP; ++e) l[e] = 0.f;
  for (int d = 0; d < DMODEL; ++d) {
    float xv = xr[d];
    const float* wr = Wr + d * NEXP;
    #pragma unroll
    for (int e = 0; e < NEXP; ++e) l[e] += xv * wr[e];
  }
  int e0 = 0;
  #pragma unroll
  for (int e = 1; e < NEXP; ++e) if (l[e] > l[e0]) e0 = e;
  int e1 = (e0 == 0) ? 1 : 0;
  #pragma unroll
  for (int e = 0; e < NEXP; ++e) if (e != e0 && l[e] > l[e1]) e1 = e;
  float z  = __expf(l[e1] - l[e0]);
  float w0 = 1.f / (1.f + z);
  float w1 = z * w0;
  int p0 = atomicAdd(&counts[e0], 1);
  tokl[e0 * NTOK + p0] = t; gatl[e0 * NTOK + p0] = w0;
  int p1 = atomicAdd(&counts[e1], 1);
  tokl[e1 * NTOK + p1] = t; gatl[e1 * NTOK + p1] = w1;
}

// fused expert FFN: 32-token tile, GEMM1 -> silu*mul (h in LDS) -> GEMM2 -> gated scatter
__global__ void __launch_bounds__(TPB) moe_ffn(
    const unsigned short* __restrict__ xbf,
    const unsigned short* __restrict__ w12b,
    const unsigned short* __restrict__ w3b,
    const int*   __restrict__ counts,
    const int*   __restrict__ tokl,
    const float* __restrict__ gatl,
    float* __restrict__ out) {
  extern __shared__ char smem_raw[];
  unsigned short* Abuf = (unsigned short*)smem_raw;                       // [BM][DMODEL]
  unsigned short* Hbuf = (unsigned short*)(smem_raw + BM * DMODEL * 2);   // [BM][HPAD]
  __shared__ int   s_tok[BM];
  __shared__ float s_gate[BM];

  const int e    = blockIdx.x / TILES;
  const int tile = blockIdx.x - e * TILES;
  const int cnt  = counts[e];
  const int m0   = tile * BM;
  if (m0 >= cnt) return;

  const int tid  = threadIdx.x;
  const int lane = tid & 31;
  const int wave = tid >> 5;
  const int mrow = lane & 15;
  const int hi   = (lane & 16) ? 8 : 0;   // A-frag K half-select
  const int moff = (lane & 16) ? 8 : 0;   // C-tile row offset

  if (tid < BM) {
    int idx = m0 + tid;
    bool v = idx < cnt;                    // invalid slots: token 0, gate 0 (adds 0.0)
    s_tok[tid]  = v ? tokl[e * NTOK + idx] : 0;
    s_gate[tid] = v ? gatl[e * NTOK + idx] : 0.f;
  }
  __syncthreads();

  // gather 32 token rows of bf16 x into LDS (16B chunks)
  for (int i = tid; i < BM * (DMODEL / 8); i += TPB) {
    int m  = i >> 7;
    int ch = (i & 127) << 3;
    *(u32x4*)(Abuf + m * DMODEL + ch) =
        *(const u32x4*)(xbf + (long long)s_tok[m] * DMODEL + ch);
  }
  // zero the K-pad columns of Hbuf (cols 2736..2751 never written by GEMM1)
  for (int i = tid; i < BM * (HPAD - HCOVER); i += TPB) {
    int m = i >> 4;
    Hbuf[m * HPAD + HCOVER + (i & 15)] = 0;
  }
  __syncthreads();

  // ---------------- GEMM1 + SiLU*mul ----------------
  const unsigned short* w12e = w12b + (long long)e * DMODEL * W12STRIDE;
  for (int jt = wave; jt < NT1; jt += NWAVE) {
    v8f acc00 = vzero8(), acc01 = vzero8(), acc10 = vzero8(), acc11 = vzero8();
    const int c1 = jt * 16;
    for (int k0 = 0; k0 < DMODEL; k0 += 32) {
      v16bf a0 = ld_frag_row(Abuf + mrow * DMODEL,        k0, hi);
      v16bf a1 = ld_frag_row(Abuf + (16 + mrow) * DMODEL, k0, hi);
      const unsigned short* brow = w12e + (long long)k0 * W12STRIDE + mrow;
      v16bf b1 = ld_frag_b(brow + c1, W12STRIDE, hi);           // gate (h1) columns
      v16bf b2 = ld_frag_b(brow + H2OFF + c1, W12STRIDE, hi);   // up   (h2) columns
      acc00 = wmma_bf16(a0, b1, acc00);
      acc10 = wmma_bf16(a1, b1, acc10);
      acc01 = wmma_bf16(a0, b2, acc01);
      acc11 = wmma_bf16(a1, b2, acc11);
    }
    #pragma unroll
    for (int r = 0; r < 8; ++r) {
      float g0 = acc00[r], u0 = acc01[r];
      float h0 = (g0 / (1.f + __expf(-g0))) * u0;
      Hbuf[(r + moff) * HPAD + c1 + mrow] = f2bf(h0);
      float g1 = acc10[r], u1 = acc11[r];
      float h1 = (g1 / (1.f + __expf(-g1))) * u1;
      Hbuf[(16 + r + moff) * HPAD + c1 + mrow] = f2bf(h1);
    }
  }
  __syncthreads();

  // ---------------- GEMM2 + gated scatter-add ----------------
  const unsigned short* w3e = w3b + (long long)e * HPAD * DMODEL;
  for (int nt = wave; nt < DMODEL / 16; nt += NWAVE) {
    v8f acc0 = vzero8(), acc1 = vzero8();
    const int c = nt * 16;
    for (int k0 = 0; k0 < HPAD; k0 += 32) {
      v16bf a0 = ld_frag_row(Hbuf + mrow * HPAD,        k0, hi);
      v16bf a1 = ld_frag_row(Hbuf + (16 + mrow) * HPAD, k0, hi);
      v16bf b  = ld_frag_b(w3e + (long long)k0 * DMODEL + c + mrow, DMODEL, hi);
      acc0 = wmma_bf16(a0, b, acc0);
      acc1 = wmma_bf16(a1, b, acc1);
    }
    #pragma unroll
    for (int r = 0; r < 8; ++r) {
      int m = r + moff;
      atomicAdd(out + (long long)s_tok[m] * DMODEL + c + mrow, acc0[r] * s_gate[m]);
      int m2 = 16 + r + moff;
      atomicAdd(out + (long long)s_tok[m2] * DMODEL + c + mrow, acc1[r] * s_gate[m2]);
    }
  }
}

// ------------------------------------------------------------------
extern "C" void kernel_launch(void* const* d_in, const int* in_sizes, int n_in,
                              void* d_out, int out_size, void* d_ws, size_t ws_size,
                              hipStream_t stream) {
  const float* x   = (const float*)d_in[0];
  const float* Wr  = (const float*)d_in[1];
  const float* W12 = (const float*)d_in[2];
  const float* W3  = (const float*)d_in[3];
  float* out = (float*)d_out;
  const int T = in_sizes[0] / DMODEL;   // 8192

  char* ws = (char*)d_ws;
  size_t off = 0;
  int*            counts = (int*)(ws + off);            off += 256;
  int*            tokl   = (int*)(ws + off);            off += (size_t)NEXP * NTOK * 4;
  float*          gatl   = (float*)(ws + off);          off += (size_t)NEXP * NTOK * 4;
  unsigned short* xbf    = (unsigned short*)(ws + off); off += (size_t)NTOK * DMODEL * 2;
  unsigned short* w12b   = (unsigned short*)(ws + off); off += (size_t)NEXP * DMODEL * W12STRIDE * 2;
  unsigned short* w3b    = (unsigned short*)(ws + off); off += (size_t)NEXP * HPAD * DMODEL * 2;
  (void)n_in; (void)ws_size; (void)out_size;

  zero_u32<<<1, 64, 0, stream>>>((unsigned*)counts, 64);
  zero_u32<<<4096, TPB, 0, stream>>>((unsigned*)out, (long long)T * DMODEL);
  conv_x  <<<4096, TPB, 0, stream>>>(x, xbf, (long long)T * DMODEL);
  conv_w12<<<8192, TPB, 0, stream>>>(W12, w12b);
  conv_w3 <<<8192, TPB, 0, stream>>>(W3, w3b);
  route   <<<(T + TPB - 1) / TPB, TPB, 0, stream>>>(x, Wr, counts, tokl, gatl, T);

  const int smem = BM * DMODEL * 2 + BM * HPAD * 2;   // 64KB + 172KB = 241664 B
  hipFuncSetAttribute(reinterpret_cast<const void*>(moe_ffn),
                      hipFuncAttributeMaxDynamicSharedMemorySize, smem);
  moe_ffn<<<NEXP * TILES, TPB, smem, stream>>>(xbf, w12b, w3b, counts, tokl, gatl, out);
}
